// SwinTransformer_5669356836350
// MI455X (gfx1250) — matrix operand order repaired and checked by reading, not compiled, not run on hardware
//
#include <hip/hip_runtime.h>

// ---------------------------------------------------------------------------
// Fused Swin window-attention block for gfx1250 (MI455X), bf16 WMMA pipeline.
// One workgroup (8 x wave32) per window; all intermediates live in LDS.
// ---------------------------------------------------------------------------

typedef __attribute__((ext_vector_type(16))) __bf16 bf16x16;
typedef __attribute__((ext_vector_type(8)))  __bf16 bf16x8;
typedef __attribute__((ext_vector_type(8)))  float  f32x8;
typedef __attribute__((ext_vector_type(4)))  float  f32x4;

#define NHEADS 8
#define HD     32
#define NTOK   49
#define NPAD   64
#define CH     256
#define NWIN   16
#define NN     (NTOK * NTOK)   // 2401

// padded LDS row strides (elements) -> conflict-free ds_load_b128 patterns
#define QS 264   // 528B row: bank step 4 for 16-lane b128 loads
#define VTS 72   // 144B row (16B aligned rows -> b128 stores legal)
#define PS 72
#define SS 65    // f32 rows, bank step 1 for 32-lane scalar sweeps

// ---- WMMA fragment loaders (contiguous k order, same for A and B) -------------
// A 16x32: lane holds row m=l&15; k = 16*half + e
__device__ __forceinline__ bf16x16 load_a16(const __bf16* row, int half) {
  union { bf16x16 v; bf16x8 h[2]; } f;
  f.h[0] = *(const bf16x8*)(row + 16 * half);
  f.h[1] = *(const bf16x8*)(row + 16 * half + 8);
  return f.v;
}
// B 32x16: lane holds col n=l&15; k = 16*half + e (caller pre-offsets 16*half)
__device__ __forceinline__ bf16x16 load_b16(const __bf16* col) {
  union { bf16x16 v; bf16x8 h[2]; } f;
  f.h[0] = *(const bf16x8*)(col);
  f.h[1] = *(const bf16x8*)(col + 8);
  return f.v;
}

__device__ __forceinline__ f32x8 wmma_bf16(bf16x16 a, bf16x16 b, f32x8 c) {
  return __builtin_amdgcn_wmma_f32_16x16x32_bf16(false, a, false, b,
                                                 (short)0, c, false, false);
}

__device__ __forceinline__ float wave_max(float v) {
#pragma unroll
  for (int off = 16; off > 0; off >>= 1) v = fmaxf(v, __shfl_xor(v, off, 32));
  return v;
}
__device__ __forceinline__ float wave_sum(float v) {
#pragma unroll
  for (int off = 16; off > 0; off >>= 1) v += __shfl_xor(v, off, 32);
  return v;
}

// ---- prep: f32->bf16 weights + per-head rel-pos-bias gather ---------------
__global__ void swin_prep(const float* __restrict__ qkv_w,
                          const float* __restrict__ proj_w,
                          const float* __restrict__ tbl,
                          const int*   __restrict__ pos_idx,
                          __bf16* __restrict__ qkv_wb,
                          __bf16* __restrict__ proj_wb,
                          float*  __restrict__ bias_full) {
  int id = blockIdx.x * blockDim.x + threadIdx.x;
  if (id < 3 * CH * CH) qkv_wb[id] = (__bf16)qkv_w[id];
  int id2 = id - 3 * CH * CH;
  if (id2 >= 0 && id2 < CH * CH) proj_wb[id2] = (__bf16)proj_w[id2];
  int id3 = id2 - CH * CH;
  if (id3 >= 0 && id3 < NHEADS * NN) {
    int h = id3 / NN;
    int r = id3 - h * NN;
    bias_full[id3] = tbl[pos_idx[r] * NHEADS + h];   // [h][m][n]
  }
}

// ---- fused main kernel: one workgroup (8 wave32) per window ---------------
__global__ __launch_bounds__(256)
void swin_fused(const float* __restrict__ x,
                const float* __restrict__ mask,
                const float* __restrict__ qkv_b,
                const float* __restrict__ proj_b,
                const __bf16* __restrict__ qkv_wb,
                const __bf16* __restrict__ proj_wb,
                const float* __restrict__ bias_full,
                float* __restrict__ out) {
  __shared__ __align__(16) __bf16 lds_q [NPAD * QS];      // Q (pre-scaled)
  __shared__ __align__(16) __bf16 lds_k [NPAD * QS];      // K
  __shared__ __align__(16) __bf16 lds_vt[CH * VTS];       // V^T: [h*32+d][m]
  __shared__ __align__(16) __bf16 lds_xo[NPAD * QS];      // x (phase1) / O
  __shared__ __align__(16) float  lds_s [NTOK * SS];      // attn logits
  __shared__ __align__(16) __bf16 lds_p [NPAD * PS];      // softmax probs

  const int b    = blockIdx.x;
  const int tid  = threadIdx.x;
  const int wave = tid >> 5;
  const int lane = tid & 31;
  const int nl   = lane & 15;
  const int half = lane >> 4;
  const float scale = 0.17677669529663689f;  // 32^-0.5

  // ---- stage x tile (rows >= 49 zero), vectorized non-temporal loads ------
  {
    const f32x4* xb4 = (const f32x4*)(x + (size_t)b * NTOK * CH);
    for (int i = tid; i < NPAD * (CH / 4); i += 256) {
      int r = i >> 6;                 // 64 f32x4 per row
      int q = i & 63;
      f32x4 v = {0.f, 0.f, 0.f, 0.f};
      if (r < NTOK) v = __builtin_nontemporal_load(xb4 + (size_t)r * 64 + q);
      __bf16* d = lds_xo + r * QS + q * 4;
      d[0] = (__bf16)v.x; d[1] = (__bf16)v.y;
      d[2] = (__bf16)v.z; d[3] = (__bf16)v.w;
    }
  }
  __syncthreads();

  // ---- phase 1: QKV = x @ qkv_w^T + b, scatter to Q/K/V^T -----------------
  // t stepping by 8 keeps mt = t&3 wave-uniform -> A frags hoisted by compiler
  for (int t = wave; t < 192; t += 8) {
    const int jt = t >> 2, mt = t & 3;
    const int j  = jt * 16 + nl;                 // output channel 0..767
    f32x8 acc = {};
    const __bf16* arow = lds_xo + (mt * 16 + nl) * QS;
    const __bf16* bcol = qkv_wb + (size_t)j * CH + 16 * half;
#pragma unroll
    for (int kk = 0; kk < 8; ++kk) {
      bf16x16 a  = load_a16(arow + kk * 32, half);
      bf16x16 bb = load_b16(bcol + kk * 32);
      acc = wmma_bf16(a, bb, acc);
    }
    const float bias  = qkv_b[j];
    const int   three = jt >> 4;                 // scalar (tile-uniform)
    const int   col   = (jt & 15) * 16 + nl;     // h*32 + d
    if (three == 0) {
#pragma unroll
      for (int r = 0; r < 8; ++r)
        lds_q[(mt * 16 + r + 8 * half) * QS + col] =
            (__bf16)((acc[r] + bias) * scale);
    } else if (three == 1) {
#pragma unroll
      for (int r = 0; r < 8; ++r)
        lds_k[(mt * 16 + r + 8 * half) * QS + col] = (__bf16)(acc[r] + bias);
    } else {
      bf16x8 pk;                                  // V^T row: 8 contiguous m
#pragma unroll
      for (int r = 0; r < 8; ++r) pk[r] = (__bf16)(acc[r] + bias);
      *(bf16x8*)(lds_vt + col * VTS + mt * 16 + 8 * half) = pk;
    }
  }
  __syncthreads();

  // ---- phase 2: per-head attention ----------------------------------------
  const int w = b & (NWIN - 1);
  for (int h = 0; h < NHEADS; ++h) {
    // S = (Q*scale) @ K^T  (hd=32 -> one bf16 WMMA K-step per tile)
    for (int t = wave; t < 16; t += 8) {
      const int mt = t >> 2, nt = t & 3;
      bf16x16 a  = load_a16(lds_q + (mt * 16 + nl) * QS + h * HD, half);
      bf16x16 bb = load_b16(lds_k + (nt * 16 + nl) * QS + h * HD + 16 * half);
      f32x8 s = {};
      s = wmma_bf16(a, bb, s);
#pragma unroll
      for (int r = 0; r < 8; ++r) {
        int m = mt * 16 + r + 8 * half;
        if (m < NTOK) lds_s[m * SS + nt * 16 + nl] = s[r];
      }
    }
    __syncthreads();

    // wave-parallel softmax: one wave per row, lanes cover cols (c, c+32)
    for (int row = wave; row < NPAD; row += 8) {
      if (row < NTOK) {
        const float* srow = lds_s + row * SS;
        const float* brow = bias_full + (size_t)h * NN + row * NTOK;
        const float* mrow = mask + ((size_t)w * NTOK + row) * NTOK;
        const int c1 = lane + 32;
        float v0 = (lane < NTOK) ? srow[lane] + brow[lane] + mrow[lane]
                                 : -3.0e38f;
        float v1 = (c1 < NTOK) ? srow[c1] + brow[c1] + mrow[c1] : -3.0e38f;
        float mx = wave_max(fmaxf(v0, v1));
        float e0 = (lane < NTOK) ? __expf(v0 - mx) : 0.f;
        float e1 = (c1 < NTOK) ? __expf(v1 - mx) : 0.f;
        float inv = 1.f / wave_sum(e0 + e1);
        __bf16* prow = lds_p + row * PS;
        prow[lane] = (__bf16)(e0 * inv);     // pad cols get exact 0
        prow[c1]   = (__bf16)(e1 * inv);
      } else {
        __bf16* prow = lds_p + row * PS;     // pad rows -> exact 0
        prow[lane] = (__bf16)0.f;
        prow[lane + 32] = (__bf16)0.f;
      }
    }
    __syncthreads();

    // O_h = P @ V   (K = 64 padded -> 2 WMMA per tile, 1 tile per wave)
    {
      const int mt = wave >> 1, dt = wave & 1;
      f32x8 o = {};
#pragma unroll
      for (int kk = 0; kk < 2; ++kk) {
        bf16x16 a  = load_a16(lds_p + (mt * 16 + nl) * PS + kk * 32, half);
        bf16x16 bb = load_b16(lds_vt + (h * HD + dt * 16 + nl) * VTS +
                              kk * 32 + 16 * half);
        o = wmma_bf16(a, bb, o);
      }
#pragma unroll
      for (int r = 0; r < 8; ++r)
        lds_xo[(mt * 16 + r + 8 * half) * QS + h * HD + dt * 16 + nl] =
            (__bf16)o[r];
    }
    __syncthreads();
  }

  // ---- phase 3: out = O @ proj_w^T + proj_b -------------------------------
  for (int t = wave; t < 64; t += 8) {
    const int nt = t >> 2, mt = t & 3;
    const int j  = nt * 16 + nl;
    f32x8 acc = {};
    const __bf16* arow = lds_xo + (mt * 16 + nl) * QS;
    const __bf16* bcol = proj_wb + (size_t)j * CH + 16 * half;
#pragma unroll
    for (int kk = 0; kk < 8; ++kk) {
      bf16x16 a  = load_a16(arow + kk * 32, half);
      bf16x16 bb = load_b16(bcol + kk * 32);
      acc = wmma_bf16(a, bb, acc);
    }
    const float pb = proj_b[j];
#pragma unroll
    for (int r = 0; r < 8; ++r) {
      int m = mt * 16 + r + 8 * half;
      if (m < NTOK)
        __builtin_nontemporal_store(acc[r] + pb,
                                    out + ((size_t)b * NTOK + m) * CH + j);
    }
  }
}

// ---------------------------------------------------------------------------
extern "C" void kernel_launch(void* const* d_in, const int* in_sizes, int n_in,
                              void* d_out, int out_size, void* d_ws,
                              size_t ws_size, hipStream_t stream) {
  const float* x       = (const float*)d_in[0];
  const float* mask    = (const float*)d_in[1];
  const float* qkv_w   = (const float*)d_in[2];
  const float* qkv_b   = (const float*)d_in[3];
  const float* proj_w  = (const float*)d_in[4];
  const float* proj_b  = (const float*)d_in[5];
  const float* tbl     = (const float*)d_in[6];
  const int*   pos_idx = (const int*)d_in[7];

  char* ws = (char*)d_ws;
  __bf16* qkv_wb    = (__bf16*)ws;                              // 393216 B
  __bf16* proj_wb   = (__bf16*)(ws + 393216);                   // 131072 B
  float*  bias_full = (float*)(ws + 393216 + 131072);           //  76832 B

  const int prep_total = 3 * CH * CH + CH * CH + NHEADS * NN;   // 281352
  swin_prep<<<(prep_total + 255) / 256, 256, 0, stream>>>(
      qkv_w, proj_w, tbl, pos_idx, qkv_wb, proj_wb, bias_full);

  swin_fused<<<2048, 256, 0, stream>>>(x, mask, qkv_b, proj_b, qkv_wb,
                                       proj_wb, bias_full, (float*)d_out);
}
